// SimpleSeqV2_6270652252905
// MI455X (gfx1250) — matrix operand order repaired and checked by reading, not compiled, not run on hardware
//
#include <hip/hip_runtime.h>
#include <hip/hip_bf16.h>

// ---------------- problem constants (match reference) ----------------
#define NN   262144        // total nodes (B * NPG)
#define EE   2097152       // total edges
#define BB   16            // graphs
#define NPG  16384         // nodes per graph
#define RR   2191          // reference-graph nodes
#define EREF (RR * 16)     // 35056 reference edges
#define HH   64            // hidden dim

typedef __attribute__((ext_vector_type(16))) _Float16 v16h;
typedef __attribute__((ext_vector_type(8)))  float    v8f;

// branch-free tanh: exp never overflows (arg <= 0), exact +-1 saturation
__device__ __forceinline__ float fast_tanh(float x) {
  float ax = fabsf(x);
  float e  = __expf(-2.0f * ax);        // v_exp_f32 path, no branches
  float r  = (1.0f - e) / (1.0f + e);
  return copysignf(r, x);
}

// ---------------- tiny utility kernels ----------------
__global__ void k_fill4(float4* __restrict__ p, int n4) {
  int i = blockIdx.x * blockDim.x + threadIdx.x;
  if (i < n4) p[i] = make_float4(0.f, 0.f, 0.f, 0.f);
}

__global__ void k_fill(float* __restrict__ p, int n, float v) {
  int i = blockIdx.x * blockDim.x + threadIdx.x;
  if (i < n) p[i] = v;
}

__global__ void k_negcopy4(const float4* __restrict__ src, float4* __restrict__ dst, int n4) {
  int i = blockIdx.x * blockDim.x + threadIdx.x;
  if (i < n4) {
    float4 v = src[i];
    dst[i] = make_float4(-v.x, -v.y, -v.z, -v.w);
  }
}

__global__ void k_negcopy(const float* __restrict__ src, float* __restrict__ dst, int n) {
  int i = blockIdx.x * blockDim.x + threadIdx.x;
  if (i < n) dst[i] = -src[i];
}

// pad [n,3] (stride 3) -> [n,32] zero-filled
__global__ void k_pad32(const float* __restrict__ src, float* __restrict__ dst, int n) {
  int i = blockIdx.x * blockDim.x + threadIdx.x;
  if (i >= n * 32) return;
  int row = i >> 5, col = i & 31;
  dst[i] = (col < 3) ? src[row * 3 + col] : 0.0f;
}

__global__ void k_degree(const int* __restrict__ rows, float* __restrict__ deg, int e) {
  int i = blockIdx.x * blockDim.x + threadIdx.x;
  if (i < e) atomicAdd(&deg[rows[i]], 1.0f);
}

__global__ void k_dinv(float* __restrict__ d, int n) {  // in place: deg -> dinv
  int i = blockIdx.x * blockDim.x + threadIdx.x;
  if (i < n) {
    float v = d[i];
    d[i] = (v > 0.0f) ? rsqrtf(fmaxf(v, 1e-12f)) : 0.0f;
  }
}

// scalar prop (used for F=3): out[row,f] += scale*(-(dinv[r]*dinv[c]))*in[col,f]
__global__ void k_prop(const int* __restrict__ rows, const int* __restrict__ cols,
                       const float* __restrict__ dinv, const float* __restrict__ in,
                       float* __restrict__ out, int e, int F, float scale) {
  long tid = (long)blockIdx.x * blockDim.x + threadIdx.x;
  if (tid >= (long)e * F) return;
  int ei = (int)(tid / F);
  int f  = (int)(tid - (long)ei * F);
  int r = rows[ei], c = cols[ei];
  float w = -(dinv[r] * dinv[c]) * scale;
  atomicAdd(&out[(long)r * F + f], w * in[(long)c * F + f]);
}

// vectorized prop for F=64: float4 gather + 4 atomics per thread (16 thr/edge)
__global__ void k_prop4(const int* __restrict__ rows, const int* __restrict__ cols,
                        const float* __restrict__ dinv, const float* __restrict__ in,
                        float* __restrict__ out, int e, float scale) {
  long tid = (long)blockIdx.x * blockDim.x + threadIdx.x;
  if (tid >= (long)e * 16) return;
  int ei = (int)(tid >> 4);
  int q  = ((int)tid & 15) * 4;
  int r = rows[ei], c = cols[ei];
  float w = -(dinv[r] * dinv[c]) * scale;
  const float4 v = *(const float4*)(in + (long)c * HH + q);
  float* o = out + (long)r * HH + q;
  atomicAdd(o + 0, w * v.x);
  atomicAdd(o + 1, w * v.y);
  atomicAdd(o + 2, w * v.z);
  atomicAdd(o + 3, w * v.w);
}

// ---------------- weight pre-pack: fragment-ready fp16 -----------------------
// out[t][kc][wave][lane][idx] = W[t, kc*32 + (idx>>3)*16 + (lane>>4)*8 + (idx&7), wave*16 + (lane&15)]
// (zero beyond Forig rows).  Matches the CDNA5 16-bit B 32x16 fragment layout.
__global__ void k_packW(const float* __restrict__ W, _Float16* __restrict__ out,
                        int Forig, int KC) {
  int j = blockIdx.x * blockDim.x + threadIdx.x;
  if (j >= 3 * KC * 2048) return;
  int idx  = j & 15;
  int lane = (j >> 4) & 31;
  int wave = (j >> 9) & 3;
  int kc   = (j >> 11) % KC;
  int t    = j / (KC * 2048);
  int k    = kc * 32 + (idx >> 3) * 16 + (lane >> 4) * 8 + (idx & 7);
  int col  = wave * 16 + (lane & 15);
  out[j] = (k < Forig) ? (_Float16)W[((long)t * Forig + k) * 64 + col] : (_Float16)0.0f;
}

// ---------------- fused ChebConv output GEMM (WMMA) ----------------
// out[n,64] = act( bias + T0@W0 + T1@W1 + T2@W2 ), A buffers stride F=KC*32,
// weights pre-packed fp16 fragments.  1 wave per 16x16 tile, 4 waves/block.
// No bounds checks in the hot loop: row clamp on loads, mask only on stores.
template <int KC, int DO_TANH>
__global__ void k_cheb_gemm_wmma(const float* __restrict__ A0,
                                 const float* __restrict__ A1v,
                                 const float* __restrict__ A2v,
                                 const _Float16* __restrict__ Wp,
                                 const float* __restrict__ bias,
                                 float* __restrict__ out,
                                 int n) {
  constexpr int F = KC * 32;
  const int lane = threadIdx.x & 31;
  const int wave = threadIdx.x >> 5;      // 0..3 -> output-col tile
  const int row0 = blockIdx.x * 16;
  const int n0   = wave * 16;
  const int lh   = lane & 15;
  const int hi   = lane >> 4;
  const int mc   = min(row0 + lh, n - 1); // clamped A row (stores are masked)

  v8f acc;
  const float bv = bias[n0 + lh];
  #pragma unroll
  for (int r = 0; r < 8; ++r) acc[r] = bv;

  #pragma unroll
  for (int t = 0; t < 3; ++t) {
    const float* __restrict__ Arow =
        (t == 0 ? A0 : (t == 1 ? A1v : A2v)) + (long)mc * F;
    #pragma unroll
    for (int kc = 0; kc < KC; ++kc) {
      const int k0 = kc * 32;
      // A 16x32 fragment: two contiguous 8-float runs per lane -> 4x b128
      const float4 a0 = *(const float4*)(Arow + k0 + hi * 8);
      const float4 a1 = *(const float4*)(Arow + k0 + hi * 8 + 4);
      const float4 a2 = *(const float4*)(Arow + k0 + 16 + hi * 8);
      const float4 a3 = *(const float4*)(Arow + k0 + 16 + hi * 8 + 4);
      v16h a;
      a[0]  = (_Float16)a0.x; a[1]  = (_Float16)a0.y;
      a[2]  = (_Float16)a0.z; a[3]  = (_Float16)a0.w;
      a[4]  = (_Float16)a1.x; a[5]  = (_Float16)a1.y;
      a[6]  = (_Float16)a1.z; a[7]  = (_Float16)a1.w;
      a[8]  = (_Float16)a2.x; a[9]  = (_Float16)a2.y;
      a[10] = (_Float16)a2.z; a[11] = (_Float16)a2.w;
      a[12] = (_Float16)a3.x; a[13] = (_Float16)a3.y;
      a[14] = (_Float16)a3.z; a[15] = (_Float16)a3.w;
      // B fragment: one 32-byte contiguous read of pre-packed fp16
      const v16h b = *(const v16h*)(Wp + ((((t * KC + kc) * 4 + wave) * 32 + lane) * 16));
      acc = __builtin_amdgcn_wmma_f32_16x16x32_f16(false, a, false, b,
                                                   (short)0, acc, false, false);
    }
  }

  #pragma unroll
  for (int r = 0; r < 8; ++r) {
    int mm = row0 + r + (hi << 3);        // C/D layout: VGPR r -> M=r+8*hi, N=lane&15
    if (mm < n) {
      float v = acc[r];
      if (DO_TANH) v = fast_tanh(v);      // branch-free epilogue
      out[(long)mm * 64 + n0 + lh] = v;
    }
  }
}

// ---------------- two-stage mean pooling -------------------------------------
__global__ void k_pool_partial(const float* __restrict__ h, float* __restrict__ pbuf) {
  // blockIdx.x in [0,32): (g,half);  blockIdx.y in [0,16): 512-node chunk
  const int g = blockIdx.x >> 1, half = blockIdx.x & 1;
  const int f = threadIdx.x;   // 64 threads
  const long base = (long)g * NPG + (long)half * (NPG / 2) + (long)blockIdx.y * 512;
  float s = 0.0f;
  for (int i = 0; i < 512; ++i) s += h[(base + i) * HH + f];
  atomicAdd(&pbuf[blockIdx.x * HH + f], s);
}

__global__ void k_pool_final(const float* __restrict__ pbuf, float* __restrict__ xbb,
                             float* __restrict__ xsc) {
  int i = blockIdx.x * blockDim.x + threadIdx.x;   // 32*64
  if (i >= 2 * BB * HH) return;
  int g = i / (2 * HH), rem = i % (2 * HH);
  int half = rem / HH, f = rem % HH;
  float mv = pbuf[(g * 2 + half) * HH + f] / (float)(NPG / 2);
  if (half == 0) xbb[g * HH + f] = mv;
  else           xsc[g * HH + f] = mv;
}

// ---------------- small dense GEMM: out[M,Nc] = (accum? out : bias) + in@W ---
__global__ void k_small_gemm(const float* __restrict__ in, const float* __restrict__ W,
                             const float* __restrict__ bias, float* __restrict__ out,
                             int M, int K, int Nc, int accum) {
  int tid = blockIdx.x * blockDim.x + threadIdx.x;
  if (tid >= M * Nc) return;
  int row = tid / Nc, col = tid - row * Nc;
  float s = accum ? out[tid] : (bias ? bias[col] : 0.0f);
  for (int k = 0; k < K; ++k) s += in[row * K + k] * W[k * Nc + col];
  out[tid] = s;
}

// ---------------- final fused MLP: relu chain over B*R rows ------------------
__global__ void k_final_mlp(const float* __restrict__ A3, const float* __restrict__ D3,
                            const float* __restrict__ W1, const float* __restrict__ b1,
                            const float* __restrict__ W2, const float* __restrict__ b2,
                            float* __restrict__ out_res) {
  __shared__ float sW1[64 * 64];
  __shared__ float sW2[64 * 3];
  __shared__ float sb1[64];
  __shared__ float sb2[3];
  for (int i = threadIdx.x; i < 64 * 64; i += blockDim.x) sW1[i] = W1[i];
  for (int i = threadIdx.x; i < 64 * 3;  i += blockDim.x) sW2[i] = W2[i];
  if (threadIdx.x < 64) sb1[threadIdx.x] = b1[threadIdx.x];
  if (threadIdx.x < 3)  sb2[threadIdx.x] = b2[threadIdx.x];
  __syncthreads();

  int row = blockIdx.x * blockDim.x + threadIdx.x;
  if (row >= BB * RR) return;
  int g = row / RR, i = row - g * RR;

  float t[64];
  #pragma unroll 4
  for (int f = 0; f < 64; ++f)
    t[f] = fmaxf(A3[i * 64 + f] + D3[g * 64 + f], 0.0f);
  float r1[64];
  for (int j = 0; j < 64; ++j) {
    float s = sb1[j];
    #pragma unroll 4
    for (int f = 0; f < 64; ++f) s += t[f] * sW1[f * 64 + j];
    r1[j] = fmaxf(s, 0.0f);
  }
  #pragma unroll
  for (int c = 0; c < 3; ++c) {
    float s = sb2[c];
    #pragma unroll 4
    for (int j = 0; j < 64; ++j) s += r1[j] * sW2[j * 3 + c];
    out_res[(long)row * 3 + c] = fmaxf(s, 0.0f);
  }
}

// ---------------- host side ----------------
static inline int cdiv(long a, long b) { return (int)((a + b - 1) / b); }

extern "C" void kernel_launch(void* const* d_in, const int* in_sizes, int n_in,
                              void* d_out, int out_size, void* d_ws, size_t ws_size,
                              hipStream_t stream) {
  (void)in_sizes; (void)n_in; (void)out_size; (void)ws_size;
  const float* x    = (const float*)d_in[0];
  const int*   eidx = (const int*)  d_in[1];
  /* d_in[2] batch, d_in[3] mask: mask structure is fixed (first half backbone) */
  const float* xref = (const float*)d_in[4];
  const int*   eref = (const int*)  d_in[5];
  const float* cw0 = (const float*)d_in[6];  const float* cb0 = (const float*)d_in[7];
  const float* cw1 = (const float*)d_in[8];  const float* cb1 = (const float*)d_in[9];
  const float* cw2 = (const float*)d_in[10]; const float* cb2 = (const float*)d_in[11];
  const float* rw0 = (const float*)d_in[12]; const float* rb0 = (const float*)d_in[13];
  const float* rw1 = (const float*)d_in[14]; const float* rb1 = (const float*)d_in[15];
  const float* rw2 = (const float*)d_in[16]; const float* rb2 = (const float*)d_in[17];
  const float* mlp_w   = (const float*)d_in[18]; const float* mlp_b   = (const float*)d_in[19];
  const float* mlpP2_w = (const float*)d_in[20]; const float* mlpP2_b = (const float*)d_in[21];
  const float* rep0_w  = (const float*)d_in[22]; const float* rep0_b  = (const float*)d_in[23];
  const float* rep1_w  = (const float*)d_in[24]; const float* rep1_b  = (const float*)d_in[25];
  const float* rep2_w  = (const float*)d_in[26]; const float* rep2_b  = (const float*)d_in[27];

  // workspace carve-up
  float* ws   = (float*)d_ws;
  float* HA   = ws;                       // N*64
  float* HB   = HA + (size_t)NN * HH;     // N*64 (also pad0/pad1 during layer0)
  float* X1   = HB + (size_t)NN * HH;     // N*64 (tail holds pad2 during layer0)
  float* P    = X1 + (size_t)NN * HH;     // N*64
  float* dinv = P  + (size_t)NN * HH;     // N
  float* rHA  = dinv + NN;                // R*64
  float* rHB  = rHA + (size_t)RR * HH;
  float* rX1  = rHB + (size_t)RR * HH;
  float* rP   = rX1 + (size_t)RR * HH;
  float* dinvR= rP  + (size_t)RR * HH;    // R
  float* A1   = dinvR + RR;               // R*64
  float* A2   = A1 + (size_t)RR * HH;
  float* A3   = A2 + (size_t)RR * HH;
  float* C1   = A3 + (size_t)RR * HH;     // B*64
  float* Dv   = C1 + BB * HH;
  float* D3   = Dv + BB * HH;
  float* pbuf = D3 + BB * HH;             // 2*B*64
  _Float16* wp = (_Float16*)(pbuf + 2 * BB * HH);
  _Float16* cwp0 = wp;            // 3*1*2048
  _Float16* cwp1 = cwp0 + 6144;   // 3*2*2048
  _Float16* cwp2 = cwp1 + 12288;
  _Float16* rwp0 = cwp2 + 12288;
  _Float16* rwp1 = rwp0 + 6144;
  _Float16* rwp2 = rwp1 + 12288;

  float* out_res = (float*)d_out;                       // [B*R, 3]
  float* out_xbb = out_res + (size_t)BB * RR * 3;       // [B, H]
  float* out_xsc = out_xbb + (size_t)BB * HH;           // [B, H]

  const int TB = 256;

  // ---- weight pre-pack (fp16 fragment layout) ----
  k_packW<<<cdiv(3 * 1 * 2048, TB), TB, 0, stream>>>(cw0, cwp0, 3,  1);
  k_packW<<<cdiv(3 * 2 * 2048, TB), TB, 0, stream>>>(cw1, cwp1, 64, 2);
  k_packW<<<cdiv(3 * 2 * 2048, TB), TB, 0, stream>>>(cw2, cwp2, 64, 2);
  k_packW<<<cdiv(3 * 1 * 2048, TB), TB, 0, stream>>>(rw0, rwp0, 3,  1);
  k_packW<<<cdiv(3 * 2 * 2048, TB), TB, 0, stream>>>(rw1, rwp1, 64, 2);
  k_packW<<<cdiv(3 * 2 * 2048, TB), TB, 0, stream>>>(rw2, rwp2, 64, 2);

  // ---- symmetric normalization weights ----
  k_fill  <<<cdiv(NN, TB), TB, 0, stream>>>(dinv, NN, 0.0f);
  k_degree<<<cdiv(EE, TB), TB, 0, stream>>>(eidx, dinv, EE);
  k_dinv  <<<cdiv(NN, TB), TB, 0, stream>>>(dinv, NN);
  k_fill  <<<cdiv(RR, TB), TB, 0, stream>>>(dinvR, RR, 0.0f);
  k_degree<<<cdiv(EREF, TB), TB, 0, stream>>>(eref, dinvR, EREF);
  k_dinv  <<<cdiv(RR, TB), TB, 0, stream>>>(dinvR, RR);

  // layer 0 (F=3 sparse path, padded-to-32 GEMM path; tanh epilogue)
  auto layer0 = [&](const float* X0, const _Float16* Wpk, const float* bias, float* OUT,
                    int n, int e, const int* rows, const int* cols, const float* dv,
                    float* bx1, float* bp, float* padbase) {
    float* pad0 = padbase;               // n*32
    float* pad1 = padbase + (size_t)n * 32;
    float* pad2 = bx1 + (size_t)n * 32;  // tail of bx1 region (bx1 uses n*3)
    k_fill   <<<cdiv(n * 3, TB), TB, 0, stream>>>(bx1, n * 3, 0.0f);
    k_prop   <<<cdiv((long)e * 3, TB), TB, 0, stream>>>(rows, cols, dv, X0, bx1, e, 3, 1.0f);
    k_negcopy<<<cdiv(n * 3, TB), TB, 0, stream>>>(X0, bp, n * 3);
    k_prop   <<<cdiv((long)e * 3, TB), TB, 0, stream>>>(rows, cols, dv, bx1, bp, e, 3, 2.0f);
    k_pad32  <<<cdiv((long)n * 32, TB), TB, 0, stream>>>(X0,  pad0, n);
    k_pad32  <<<cdiv((long)n * 32, TB), TB, 0, stream>>>(bx1, pad1, n);
    k_pad32  <<<cdiv((long)n * 32, TB), TB, 0, stream>>>(bp,  pad2, n);
    k_cheb_gemm_wmma<1, 1><<<cdiv(n, 16), 128, 0, stream>>>(pad0, pad1, pad2, Wpk, bias, OUT, n);
  };
  // layers 1/2 (F=64)
  auto layer64 = [&](const float* X0, const _Float16* Wpk, const float* bias, float* OUT,
                     int n, int e, const int* rows, const int* cols, const float* dv,
                     float* bx1, float* bp, int do_tanh) {
    const int n4 = n * HH / 4;
    k_fill4   <<<cdiv(n4, TB), TB, 0, stream>>>((float4*)bx1, n4);
    k_prop4   <<<cdiv((long)e * 16, TB), TB, 0, stream>>>(rows, cols, dv, X0, bx1, e, 1.0f);
    k_negcopy4<<<cdiv(n4, TB), TB, 0, stream>>>((const float4*)X0, (float4*)bp, n4);
    k_prop4   <<<cdiv((long)e * 16, TB), TB, 0, stream>>>(rows, cols, dv, bx1, bp, e, 2.0f);
    if (do_tanh)
      k_cheb_gemm_wmma<2, 1><<<cdiv(n, 16), 128, 0, stream>>>(X0, bx1, bp, Wpk, bias, OUT, n);
    else
      k_cheb_gemm_wmma<2, 0><<<cdiv(n, 16), 128, 0, stream>>>(X0, bx1, bp, Wpk, bias, OUT, n);
  };

  // ---- main GNN stack ----
  layer0 (x,  cwp0, cb0, HA, NN, EE, eidx, eidx + EE, dinv, X1, P, HB);
  layer64(HA, cwp1, cb1, HB, NN, EE, eidx, eidx + EE, dinv, X1, P, 1);
  layer64(HB, cwp2, cb2, HA, NN, EE, eidx, eidx + EE, dinv, X1, P, 0);

  // ---- pooling (x_bb / x_sc written straight into d_out) ----
  k_fill<<<cdiv(2 * BB * HH, TB), TB, 0, stream>>>(pbuf, 2 * BB * HH, 0.0f);
  k_pool_partial<<<dim3(2 * BB, 16), HH, 0, stream>>>(HA, pbuf);
  k_pool_final<<<cdiv(2 * BB * HH, TB), TB, 0, stream>>>(pbuf, out_xbb, out_xsc);

  // ---- reference-graph stack (tanh after every layer per source bug) ----
  layer0 (xref, rwp0, rb0, rHA, RR, EREF, eref, eref + EREF, dinvR, rX1, rP, rHB);
  layer64(rHA,  rwp1, rb1, rHB, RR, EREF, eref, eref + EREF, dinvR, rX1, rP, 1);
  layer64(rHB,  rwp2, rb2, rHA, RR, EREF, eref, eref + EREF, dinvR, rX1, rP, 1);

  // ---- low-rank decomposition of phase1/phase2/rep0 ----
  k_small_gemm<<<cdiv((long)RR * HH, TB), TB, 0, stream>>>(rHA, mlp_w,             nullptr, A1, RR, HH, HH, 0);
  k_small_gemm<<<cdiv((long)BB * HH, TB), TB, 0, stream>>>(out_xbb, mlp_w + HH*HH, mlp_b,   C1, BB, HH, HH, 0);
  k_small_gemm<<<cdiv((long)RR * HH, TB), TB, 0, stream>>>(A1, mlpP2_w,            nullptr, A2, RR, HH, HH, 0);
  k_small_gemm<<<cdiv((long)BB * HH, TB), TB, 0, stream>>>(C1, mlpP2_w,            mlpP2_b, Dv, BB, HH, HH, 0);
  k_small_gemm<<<cdiv((long)BB * HH, TB), TB, 0, stream>>>(out_xsc, mlpP2_w + HH*HH, nullptr, Dv, BB, HH, HH, 1);
  k_small_gemm<<<cdiv((long)RR * HH, TB), TB, 0, stream>>>(A2, rep0_w,             nullptr, A3, RR, HH, HH, 0);
  k_small_gemm<<<cdiv((long)BB * HH, TB), TB, 0, stream>>>(Dv, rep0_w,             rep0_b,  D3, BB, HH, HH, 0);

  // ---- final ReLU MLP over all B*R residue rows -> res ----
  k_final_mlp<<<cdiv((long)BB * RR, 128), 128, 0, stream>>>(A3, D3, rep1_w, rep1_b,
                                                            rep2_w, rep2_b, out_res);
}
